// TextGenerator_1692217115006
// MI455X (gfx1250) — matrix operand order repaired and checked by reading, not compiled, not run on hardware
//
#include <hip/hip_runtime.h>
#include <cstdint>
#include <cstddef>

// ---- problem constants ----
#define TB 64
#define TL 32
#define TT 31          // L-1 timesteps
#define TD 768
#define THH 768
#define TG 2304        // 3*H
#define TV 30522
#define NROWS (TB*TT)  // 1984 (= 124*16 = 31*64, exact)

// vocab tiling: workgroup = 8 waves, 64 rows x 256 cols, K-step 32
#define VCH_COLS 256
#define NCH2 ((TV + VCH_COLS - 1)/VCH_COLS)   // 120 column chunks
#define NSUB  (NCH2 * 8)                      // 960 per-row partials (32 cols each)
#define NKSTEP (TD/32)                        // 24

typedef __attribute__((ext_vector_type(16))) __bf16 v16bf;
typedef __attribute__((ext_vector_type(8)))  float  v8f;

struct U4 { unsigned int x, y, z, w; };
union ABreg { v16bf v; U4 q[2]; };

__device__ __forceinline__ unsigned short f32_to_bf16(float f) {
  union { float f; unsigned int u; } v; v.f = f;
  unsigned int u = v.u;
  u += 0x7FFFu + ((u >> 16) & 1u);      // round-to-nearest-even
  return (unsigned short)(u >> 16);
}

// async 16B copy global -> LDS (CDNA5, ASYNCcnt-tracked)
__device__ __forceinline__ void async_copy16(unsigned lds_off, const void* gptr) {
  asm volatile("global_load_async_to_lds_b128 %0, %1, off"
               :: "v"(lds_off), "v"(gptr) : "memory");
}
__device__ __forceinline__ void wait_async_le5() {
  asm volatile("s_wait_asynccnt 0x5" ::: "memory");
}
__device__ __forceinline__ void wait_async_0() {
  asm volatile("s_wait_asynccnt 0x0" ::: "memory");
}

// ---------------- conversion kernels ----------------
__global__ void convert_bf16_kernel(const float* __restrict__ src,
                                    unsigned short* __restrict__ dst, long n) {
  long i = (long)blockIdx.x * blockDim.x + threadIdx.x;
  if (i < n) dst[i] = f32_to_bf16(src[i]);
}

// embedding[:, :-1, :] -> Xb rows r=b*31+t
__global__ void convert_emb_kernel(const float* __restrict__ emb,
                                   unsigned short* __restrict__ dst) {
  long i = (long)blockIdx.x * blockDim.x + threadIdx.x;
  if (i >= (long)NROWS * TD) return;
  int r = (int)(i / TD), d = (int)(i % TD);
  int b = r / TT, t = r % TT;
  dst[i] = f32_to_bf16(emb[((size_t)b * TL + t) * TD + d]);
}

__global__ void init_h_kernel(const float* __restrict__ img,
                              float* __restrict__ h,
                              unsigned short* __restrict__ hb) {
  int i = blockIdx.x * blockDim.x + threadIdx.x;
  if (i >= TB * THH) return;
  float v = img[i];
  h[i] = v;
  hb[i] = f32_to_bf16(v);
}

// ---------------- generic bf16-WMMA GEMM:  C = A[M,K] * B[N,K]^T + bias ----------------
// one wave (32 threads) computes a 16x16 tile; grid = (M/16, N/16)
__global__ void gemm_bias_kernel(const unsigned short* __restrict__ A,
                                 const unsigned short* __restrict__ Bm,
                                 const float* __restrict__ bias,
                                 float* __restrict__ Cf,
                                 unsigned short* __restrict__ Cb,   // optional
                                 int N, int K,
                                 int act, const float* __restrict__ alpha_p) {
  const int l    = threadIdx.x;
  const int m0   = blockIdx.x * 16;
  const int n0   = blockIdx.y * 16;
  const int half = l >> 4;
  const int lm   = l & 15;

  const unsigned short* arow = A  + (size_t)(m0 + lm) * K;
  const unsigned short* brow = Bm + (size_t)(n0 + lm) * K;

  v8f c = {};
  for (int k = 0; k < K; k += 32) {
    ABreg a, b;
    a.q[0] = *(const U4*)(arow + k + half * 8);
    a.q[1] = *(const U4*)(arow + k + 16 + half * 8);
    b.q[0] = *(const U4*)(brow + k + half * 8);
    b.q[1] = *(const U4*)(brow + k + 16 + half * 8);
    c = __builtin_amdgcn_wmma_f32_16x16x32_bf16(false, a.v, false, b.v,
                                                (short)0, c, false, false);
  }

  const int col   = n0 + lm;
  const int rbase = m0 + half * 8;
  const float bv  = bias ? bias[col] : 0.0f;
  const float al  = (act == 1 && alpha_p) ? alpha_p[0] : 0.0f;
#pragma unroll
  for (int i = 0; i < 8; i++) {
    float v = c[i] + bv;
    if (act == 1) v = (v >= 0.0f) ? v : al * v;   // PReLU
    size_t off = (size_t)(rbase + i) * N + col;
    Cf[off] = v;
    if (Cb) Cb[off] = f32_to_bf16(v);
  }
}

// ---------------- GRU gate fusion (one timestep) ----------------
__global__ void gru_gate_kernel(const float* __restrict__ gi,
                                const float* __restrict__ gh,
                                float* __restrict__ h,
                                unsigned short* __restrict__ hb,
                                unsigned short* __restrict__ hsb,
                                int t) {
  int i = blockIdx.x * blockDim.x + threadIdx.x;
  if (i >= TB * THH) return;
  int b = i / THH, j = i % THH;
  int row = b * TT + t;
  size_t gir = (size_t)row * TG;
  size_t ghr = (size_t)b * TG;
  float ir = gi[gir + j], iz = gi[gir + THH + j], in_ = gi[gir + 2 * THH + j];
  float hr = gh[ghr + j], hz = gh[ghr + THH + j], hn  = gh[ghr + 2 * THH + j];
  float r = 1.0f / (1.0f + __expf(-(ir + hr)));
  float z = 1.0f / (1.0f + __expf(-(iz + hz)));
  float n = tanhf(in_ + r * hn);
  float hv = h[i];
  float hnew = (1.0f - z) * n + z * hv;
  h[i]  = hnew;
  hb[i] = f32_to_bf16(hnew);
  hsb[(size_t)row * THH + j] = f32_to_bf16(hnew);
}

// ---------------- fused vocab GEMM + online softmax partials ----------------
// block = 8 waves, tile = 64 rows x 256 cols. A (64x32) and B (256x32) staged
// into LDS per K-step via async global->LDS copies, double-buffered.
// Wave w computes 4 M-tiles x 2 N-tiles (cols [w*32, w*32+32)).
__global__ void __launch_bounds__(256)
vocab_kernel(const unsigned short* __restrict__ X,   // [1984,768] bf16
             const unsigned short* __restrict__ Wb,  // [V,768]    bf16
             float* __restrict__ pmax,
             float* __restrict__ psum) {
  __shared__ unsigned short lA[2][64 * 32];    //  8 KB
  __shared__ unsigned short lB[2][256 * 32];   // 32 KB

  const int tid  = threadIdx.x;
  const int w    = tid >> 5;        // wave 0..7
  const int l    = tid & 31;
  const int half = l >> 4;
  const int lm   = l & 15;
  const int m0   = blockIdx.x * 64;
  const int c0   = blockIdx.y * VCH_COLS;

  // ---- per-thread copy plan: 5 x 16B chunks per K-step ----
  // chunks 0..255   -> A rows (64 rows x 4 quarters)
  // chunks 256..1279-> B rows (256 cols x 4 quarters), clamped at V
  const unsigned short* gsrc[5];
  unsigned loff0[5], loff1[5];
#pragma unroll
  for (int j = 0; j < 5; j++) {
    int c = tid + j * 256;
    if (c < 256) {
      int row = c >> 2, q = c & 3;
      gsrc[j]  = X + (size_t)(m0 + row) * TD + q * 8;
      loff0[j] = (unsigned)(uintptr_t)&lA[0][row * 32 + q * 8];
      loff1[j] = (unsigned)(uintptr_t)&lA[1][row * 32 + q * 8];
    } else {
      int c2 = c - 256;
      int n = c2 >> 2, q = c2 & 3;
      int ncol = c0 + n; if (ncol > TV - 1) ncol = TV - 1;   // clamp; masked later
      gsrc[j]  = Wb + (size_t)ncol * TD + q * 8;
      loff0[j] = (unsigned)(uintptr_t)&lB[0][n * 32 + q * 8];
      loff1[j] = (unsigned)(uintptr_t)&lB[1][n * 32 + q * 8];
    }
  }

  v8f zero = {};
  v8f acc[4][2];
#pragma unroll
  for (int mt = 0; mt < 4; mt++)
#pragma unroll
    for (int nt = 0; nt < 2; nt++) acc[mt][nt] = zero;

  // prologue: stage 0 into buffer 0
#pragma unroll
  for (int j = 0; j < 5; j++) async_copy16(loff0[j], gsrc[j]);

  for (int s = 0; s < NKSTEP; s++) {
    const int buf = s & 1;
    if (s + 1 < NKSTEP) {
      const int k0 = (s + 1) * 32;
#pragma unroll
      for (int j = 0; j < 5; j++)
        async_copy16(((s + 1) & 1) ? loff1[j] : loff0[j], gsrc[j] + k0);
      wait_async_le5();          // stage-s copies (this wave) retired
    } else {
      wait_async_0();
    }
    __syncthreads();             // all waves' stage-s data visible

    // load fragments from LDS and multiply
    ABreg a[4], b[2];
#pragma unroll
    for (int mt = 0; mt < 4; mt++) {
      const unsigned short* ar = &lA[buf][(mt * 16 + lm) * 32];
      a[mt].q[0] = *(const U4*)(ar + half * 8);
      a[mt].q[1] = *(const U4*)(ar + 16 + half * 8);
    }
#pragma unroll
    for (int nt = 0; nt < 2; nt++) {
      const unsigned short* br = &lB[buf][(w * 32 + nt * 16 + lm) * 32];
      b[nt].q[0] = *(const U4*)(br + half * 8);
      b[nt].q[1] = *(const U4*)(br + 16 + half * 8);
    }
#pragma unroll
    for (int mt = 0; mt < 4; mt++)
#pragma unroll
      for (int nt = 0; nt < 2; nt++)
        acc[mt][nt] = __builtin_amdgcn_wmma_f32_16x16x32_bf16(
            false, a[mt].v, false, b[nt].v, (short)0, acc[mt][nt], false, false);

    __syncthreads();             // done reading buf before it is overwritten
  }

  // ---- per-row online-softmax partials over this wave's 32 columns ----
  const int sub = blockIdx.y * 8 + w;          // subchunk id (32 cols)
#pragma unroll
  for (int mt = 0; mt < 4; mt++) {
    float lmax[8], lsum[8];
#pragma unroll
    for (int i = 0; i < 8; i++) lmax[i] = -3.0e38f;
#pragma unroll
    for (int nt = 0; nt < 2; nt++) {
      bool ok = (c0 + w * 32 + nt * 16 + lm) < TV;
#pragma unroll
      for (int i = 0; i < 8; i++)
        if (ok) lmax[i] = fmaxf(lmax[i], acc[mt][nt][i]);
    }
#pragma unroll
    for (int sft = 1; sft < 16; sft <<= 1)
#pragma unroll
      for (int i = 0; i < 8; i++)
        lmax[i] = fmaxf(lmax[i], __shfl_xor(lmax[i], sft, 32));

#pragma unroll
    for (int i = 0; i < 8; i++) lsum[i] = 0.0f;
#pragma unroll
    for (int nt = 0; nt < 2; nt++) {
      bool ok = (c0 + w * 32 + nt * 16 + lm) < TV;
#pragma unroll
      for (int i = 0; i < 8; i++)
        if (ok) lsum[i] += __expf(acc[mt][nt][i] - lmax[i]);
    }
#pragma unroll
    for (int sft = 1; sft < 16; sft <<= 1)
#pragma unroll
      for (int i = 0; i < 8; i++)
        lsum[i] += __shfl_xor(lsum[i], sft, 32);

    if (lm == 0) {
#pragma unroll
      for (int i = 0; i < 8; i++) {
        int row = m0 + mt * 16 + half * 8 + i;
        pmax[(size_t)row * NSUB + sub] = lmax[i];
        psum[(size_t)row * NSUB + sub] = lsum[i];
      }
    }
  }
}

// ---------------- target logit (fp32 dot), one wave per row ----------------
__global__ void tlogit_kernel(const float* __restrict__ x2f,
                              const float* __restrict__ W,
                              const int* __restrict__ idx,
                              float* __restrict__ tlog) {
  int r = blockIdx.x, l = threadIdx.x;
  int b = r / TT, t = r % TT;
  int target = idx[b * TL + t + 1];
  const float* x = x2f + (size_t)r * TD;
  const float* w = W + (size_t)target * TD;
  float s = 0.0f;
  for (int j = l; j < TD; j += 32) s += x[j] * w[j];
#pragma unroll
  for (int sft = 16; sft >= 1; sft >>= 1) s += __shfl_xor(s, sft, 32);
  if (l == 0) tlog[r] = s;
}

// ---------------- merge subchunk partials -> row max / row sumexp ----------------
__global__ void merge_kernel(const float* __restrict__ pmax,
                             const float* __restrict__ psum,
                             float* __restrict__ rmax,
                             float* __restrict__ rsum) {
  int r = blockIdx.x * blockDim.x + threadIdx.x;
  if (r >= NROWS) return;
  const float* pm = pmax + (size_t)r * NSUB;
  const float* ps = psum + (size_t)r * NSUB;
  float M = -3.0e38f;
  for (int c = 0; c < NSUB; c++) M = fmaxf(M, pm[c]);
  float S = 0.0f;
  for (int c = 0; c < NSUB; c++) S += ps[c] * __expf(pm[c] - M);
  rmax[r] = M;
  rsum[r] = S;
}

// ---------------- final loss reduction (single block, deterministic) ----------------
__global__ void loss_kernel(const int* __restrict__ idx,
                            const float* __restrict__ tlog,
                            const float* __restrict__ rmax,
                            const float* __restrict__ rsum,
                            float* __restrict__ out) {
  __shared__ float sm[256];
  float acc = 0.0f;
  for (int r = threadIdx.x; r < NROWS; r += 256) {
    int b = r / TT, t = r % TT;
    int target = idx[b * TL + t + 1];
    if (target != 0) {
      float lp = tlog[r] - rmax[r] - logf(rsum[r]);
      acc += -lp;
    }
  }
  sm[threadIdx.x] = acc;
  __syncthreads();
  for (int s = 128; s > 0; s >>= 1) {
    if (threadIdx.x < (unsigned)s) sm[threadIdx.x] += sm[threadIdx.x + s];
    __syncthreads();
  }
  if (threadIdx.x == 0) out[0] = sm[0] / (float)TB;
}

// ---------------- host launcher ----------------
extern "C" void kernel_launch(void* const* d_in, const int* in_sizes, int n_in,
                              void* d_out, int out_size, void* d_ws, size_t ws_size,
                              hipStream_t stream) {
  (void)in_sizes; (void)n_in; (void)out_size; (void)ws_size;

  const int*   idx   = (const int*)  d_in[0];
  const float* emb   = (const float*)d_in[1];
  const float* img   = (const float*)d_in[2];
  const float* W     = (const float*)d_in[3];
  const float* w_ih  = (const float*)d_in[4];
  const float* w_hh  = (const float*)d_in[5];
  const float* b_ih  = (const float*)d_in[6];
  const float* b_hh  = (const float*)d_in[7];
  const float* w1    = (const float*)d_in[8];
  const float* b1    = (const float*)d_in[9];
  const float* prelu = (const float*)d_in[10];
  const float* w2    = (const float*)d_in[11];
  const float* b2    = (const float*)d_in[12];
  float* out = (float*)d_out;

  // carve workspace (all 256B aligned)
  char* wp = (char*)d_ws;
  auto alloc = [&](size_t bytes) -> void* {
    void* p = (void*)wp;
    wp += (bytes + 255) & ~(size_t)255;
    return p;
  };
  unsigned short* Wb   = (unsigned short*)alloc((size_t)TV * TD * 2);
  unsigned short* wihb = (unsigned short*)alloc((size_t)TG * TD * 2);
  unsigned short* whhb = (unsigned short*)alloc((size_t)TG * TD * 2);
  unsigned short* w1b  = (unsigned short*)alloc((size_t)THH * TD * 2);
  unsigned short* w2b  = (unsigned short*)alloc((size_t)THH * TD * 2);
  unsigned short* Xb   = (unsigned short*)alloc((size_t)NROWS * TD * 2);
  float*          gi   = (float*)alloc((size_t)NROWS * TG * 4);
  float*          gh   = (float*)alloc((size_t)TB * TG * 4);
  float*          hf   = (float*)alloc((size_t)TB * THH * 4);
  unsigned short* hb   = (unsigned short*)alloc((size_t)TB * THH * 2);
  unsigned short* hsb  = (unsigned short*)alloc((size_t)NROWS * THH * 2);
  float*          x1f  = (float*)alloc((size_t)NROWS * THH * 4);
  unsigned short* x1b  = (unsigned short*)alloc((size_t)NROWS * THH * 2);
  float*          x2f  = (float*)alloc((size_t)NROWS * THH * 4);
  unsigned short* x2b  = (unsigned short*)alloc((size_t)NROWS * THH * 2);
  float*          pmax = (float*)alloc((size_t)NROWS * NSUB * 4);
  float*          psum = (float*)alloc((size_t)NROWS * NSUB * 4);
  float*          rmax = (float*)alloc((size_t)NROWS * 4);
  float*          rsum = (float*)alloc((size_t)NROWS * 4);
  float*          tlog = (float*)alloc((size_t)NROWS * 4);

  auto cdiv = [](long a, long b) { return (int)((a + b - 1) / b); };

  // 1) precision conversions (bf16 for all WMMA operands)
  convert_bf16_kernel<<<cdiv((long)TV * TD, 256), 256, 0, stream>>>(W, Wb, (long)TV * TD);
  convert_bf16_kernel<<<cdiv((long)TG * TD, 256), 256, 0, stream>>>(w_ih, wihb, (long)TG * TD);
  convert_bf16_kernel<<<cdiv((long)TG * TD, 256), 256, 0, stream>>>(w_hh, whhb, (long)TG * TD);
  convert_bf16_kernel<<<cdiv((long)THH * TD, 256), 256, 0, stream>>>(w1, w1b, (long)THH * TD);
  convert_bf16_kernel<<<cdiv((long)THH * TD, 256), 256, 0, stream>>>(w2, w2b, (long)THH * TD);
  convert_emb_kernel<<<cdiv((long)NROWS * TD, 256), 256, 0, stream>>>(emb, Xb);
  init_h_kernel<<<cdiv(TB * THH, 256), 256, 0, stream>>>(img, hf, hb);

  // 2) gi = X @ w_ih^T + b_ih   [1984 x 2304]
  gemm_bias_kernel<<<dim3(NROWS / 16, TG / 16), 32, 0, stream>>>(
      Xb, wihb, b_ih, gi, nullptr, TG, TD, 0, nullptr);

  // 3) GRU recurrence (sequential over 31 steps)
  for (int t = 0; t < TT; t++) {
    gemm_bias_kernel<<<dim3(TB / 16, TG / 16), 32, 0, stream>>>(
        hb, whhb, b_hh, gh, nullptr, TG, TD, 0, nullptr);
    gru_gate_kernel<<<cdiv(TB * THH, 256), 256, 0, stream>>>(gi, gh, hf, hb, hsb, t);
  }

  // 4) x1 = PReLU(hs @ w1^T + b1); x2 = x1 @ w2^T + b2
  gemm_bias_kernel<<<dim3(NROWS / 16, THH / 16), 32, 0, stream>>>(
      hsb, w1b, b1, x1f, x1b, THH, TD, 1, prelu);
  gemm_bias_kernel<<<dim3(NROWS / 16, THH / 16), 32, 0, stream>>>(
      x1b, w2b, b2, x2f, x2b, THH, TD, 0, nullptr);

  // 5) fused vocab GEMM + online softmax partials (LDS-tiled, async copies)
  vocab_kernel<<<dim3(NROWS / 64, NCH2), 256, 0, stream>>>(x2b, Wb, pmax, psum);

  // 6) target logits (fp32), merge partials, final loss
  tlogit_kernel<<<NROWS, 32, 0, stream>>>(x2f, W, idx, tlog);
  merge_kernel<<<cdiv(NROWS, 256), 256, 0, stream>>>(pmax, psum, rmax, rsum);
  loss_kernel<<<1, 256, 0, stream>>>(idx, tlog, rmax, rsum, out);
}
